// DenseFeatureExtractionModule_AP_OS4_82815559401760
// MI455X (gfx1250) — compile-verified, matrix operand order checked by reading, not run on hardware
//
#include <hip/hip_runtime.h>
#include <hip/hip_bf16.h>
#include <stdint.h>

// ---------------------------------------------------------------------------
// Types for CDNA5 WMMA (wave32): 16x16x32 bf16 -> f32 accumulate
// ---------------------------------------------------------------------------
typedef __attribute__((ext_vector_type(16))) __bf16 v16bf;
typedef __attribute__((ext_vector_type(8)))  __bf16 v8bf;
typedef __attribute__((ext_vector_type(8)))  float  v8f;

#define PAD 2           // halo width (covers dilation 2)
#define PAD2 4          // both sides

static __device__ __forceinline__ unsigned short f32_to_bf16_bits(float f) {
  union { float f; unsigned int u; } c; c.f = f;
  unsigned int u = c.u;
  unsigned int r = u + 0x7FFFu + ((u >> 16) & 1u);     // round-to-nearest-even
  if ((u & 0x7F800000u) == 0x7F800000u) r = u;         // inf/nan passthrough
  return (unsigned short)(r >> 16);
}
static __device__ __forceinline__ float bf16_bits_to_f32(unsigned short h) {
  union { unsigned int u; float f; } c; c.u = ((unsigned int)h) << 16;
  return c.f;
}
static __device__ __forceinline__ __bf16 bits_to_bf(unsigned short h) {
  union { unsigned short u; __bf16 b; } c; c.u = h; return c.b;
}

// ---------------------------------------------------------------------------
// Zero-fill (b128 stores). n8 = element count / 8.
// ---------------------------------------------------------------------------
__global__ void zero_u16x8(unsigned short* __restrict__ p, size_t n8) {
  const size_t i = (size_t)blockIdx.x * blockDim.x + threadIdx.x;
  if (i >= n8) return;
  reinterpret_cast<uint4*>(p)[i] = make_uint4(0u, 0u, 0u, 0u);
}

// ---------------------------------------------------------------------------
// Weight repack: OIHW f32 [Cout][Cin][3][3] -> [Cout][9][Cin] bf16.
// ---------------------------------------------------------------------------
__global__ void repack_w_bf16(const float* __restrict__ in,
                              unsigned short* __restrict__ out,
                              int Cin, int total) {
  int i = blockIdx.x * blockDim.x + threadIdx.x;
  if (i >= total) return;
  int ci  = i % Cin;
  int t   = i / Cin;
  int tap = t % 9;
  int co  = t / 9;
  out[i] = f32_to_bf16_bits(in[(size_t)co * Cin * 9 + (size_t)ci * 9 + tap]);
}

// ---------------------------------------------------------------------------
// Layer 1 (Cin=3): direct conv, NCHW f32 in -> padded NHWC bf16 out.
// ---------------------------------------------------------------------------
__global__ void conv1_direct(const float* __restrict__ in,    // [B][3][H][W]
                             const float* __restrict__ w,     // [64][3][3][3]
                             const float* __restrict__ bias,  // [64]
                             unsigned short* __restrict__ out,// padded NHWC C=64
                             int H, int W) {
  const size_t total = (size_t)4 * H * W * 64;
  const size_t i = (size_t)blockIdx.x * blockDim.x + threadIdx.x;
  if (i >= total) return;
  const int co  = (int)(i & 63);
  size_t t = i >> 6;
  const int pix = (int)(t % ((size_t)H * W));
  const int b   = (int)(t / ((size_t)H * W));
  const int py = pix / W;
  const int px = pix - py * W;
  float acc = bias[co];
  #pragma unroll
  for (int ci = 0; ci < 3; ++ci) {
    #pragma unroll
    for (int r = 0; r < 3; ++r) {
      #pragma unroll
      for (int s = 0; s < 3; ++s) {
        const int iy = py + r - 1, ix = px + s - 1;
        if (iy >= 0 && iy < H && ix >= 0 && ix < W)
          acc += in[(((size_t)b * 3 + ci) * H + iy) * W + ix] *
                 w[co * 27 + ci * 9 + r * 3 + s];
      }
    }
  }
  acc = acc > 0.0f ? acc : 0.0f;
  const int Wp = W + PAD2;
  out[(((size_t)b * (H + PAD2) + py + PAD) * Wp + px + PAD) * 64 + co] =
      f32_to_bf16_bits(acc);
}

// ---------------------------------------------------------------------------
// Implicit-GEMM 3x3 conv via V_WMMA_F32_16X16X32_BF16 on zero-padded NHWC.
//   Per wave: M = 64 couts (4 tiles) x N = 32 pixels (2 tiles), K = 9*Cin,
//   tap-major (Cin multiple of 32). Per-pixel dilation 1/2 via mask.
//   Branch-free inner loop: the 2-pixel zero halo makes every tap load valid.
//   grid = (H*W/32, Cout/64, B), block = 32 (one wave).
// Fragment striping (16-bit, CDNA5 ISA): lane row/col = lane&15,
//   khalf = lane>>4, element e -> k-offset 16*(e>>3)+8*khalf+(e&7): two
//   contiguous 16B runs -> two b128 loads per fragment.
// ---------------------------------------------------------------------------
__global__ __launch_bounds__(32) void conv3x3_wmma_nhwc(
    const unsigned short* __restrict__ x,   // padded [B][H+4][W+4][Cin]
    const unsigned short* __restrict__ w,   // [Cout][9][Cin] bf16
    const float* __restrict__ bias,         // [Cout]
    unsigned short* __restrict__ ybf,       // padded [B][H+4][W+4][Cout] or null
    float* __restrict__ yf32,               // [B][Cout][H][W] f32 or null
    const int* __restrict__ mask,           // [B][H/2][W/2] int or null
    int Cin, int Cout, int H, int W)
{
  const int lane  = threadIdx.x;
  const int n     = lane & 15;
  const int khalf = lane >> 4;
  const int koff  = khalf << 3;             // 0 or 8 elements
  const int b     = blockIdx.z;
  const int coBase  = blockIdx.y << 6;      // 64 couts per wave
  const int pixBase = blockIdx.x << 5;      // 32 pixels per wave (one row)
  const int py = pixBase / W;               // wave-uniform row
  const int Wp = W + PAD2;

  int px[2], dil[2];
  const unsigned short* p0[2];
  const unsigned short* xrow = x + (size_t)b * (H + PAD2) * Wp * Cin;
  #pragma unroll
  for (int u = 0; u < 2; ++u) {
    const int pix = pixBase + 16 * u + n;
    px[u] = pix - py * W;
    dil[u] = 1;
    if (mask) {
      const int m = mask[((size_t)b * (H >> 1) + (py >> 1)) * (W >> 1)
                         + (px[u] >> 1)];
      dil[u] = 1 + (m != 0);
    }
    p0[u] = xrow + ((size_t)(py + PAD) * Wp + px[u] + PAD) * Cin;
  }

  const int K9 = 9 * Cin;
  const unsigned short* wrow[4];
  #pragma unroll
  for (int t = 0; t < 4; ++t) {
    wrow[t] = w + (size_t)(coBase + 16 * t + n) * K9;
    __builtin_prefetch(wrow[t], 0, 0);       // global_prefetch_b8 (L2-resident)
  }

  v8f acc[8];                                // [t][u] -> acc[2*t+u]
  #pragma unroll
  for (int a = 0; a < 8; ++a)
    #pragma unroll
    for (int r = 0; r < 8; ++r) acc[a][r] = 0.0f;

  static const int DR[9] = {-1,-1,-1, 0,0,0, 1,1,1};
  static const int DS[9] = {-1, 0, 1,-1,0,1,-1,0,1};

  #pragma unroll
  for (int tap = 0; tap < 9; ++tap) {
    // Per-lane byte-free element offset into the padded image (can be <0).
    const int tapStep = DR[tap] * Wp + DS[tap];
    const unsigned short* bbase[2];
    #pragma unroll
    for (int u = 0; u < 2; ++u)
      bbase[u] = p0[u] + (ptrdiff_t)(dil[u] * tapStep) * Cin;
    const int tapOff = tap * Cin;

    for (int cb = 0; cb < Cin; cb += 32) {
      // ---- issue all 12 b128 loads, then 8 WMMAs ----
      v16bf bfr[2], afr[4];
      #pragma unroll
      for (int u = 0; u < 2; ++u) {
        const v8bf lo = *reinterpret_cast<const v8bf*>(bbase[u] + cb + koff);
        const v8bf hi = *reinterpret_cast<const v8bf*>(bbase[u] + cb + 16 + koff);
        bfr[u] = __builtin_shufflevector(lo, hi, 0,1,2,3,4,5,6,7,
                                                 8,9,10,11,12,13,14,15);
      }
      #pragma unroll
      for (int t = 0; t < 4; ++t) {
        const unsigned short* ap = wrow[t] + tapOff + cb;
        const v8bf alo = *reinterpret_cast<const v8bf*>(ap + koff);
        const v8bf ahi = *reinterpret_cast<const v8bf*>(ap + 16 + koff);
        afr[t] = __builtin_shufflevector(alo, ahi, 0,1,2,3,4,5,6,7,
                                                   8,9,10,11,12,13,14,15);
      }
      #pragma unroll
      for (int t = 0; t < 4; ++t)
        #pragma unroll
        for (int u = 0; u < 2; ++u)
          acc[2 * t + u] = __builtin_amdgcn_wmma_f32_16x16x32_bf16(
              false, afr[t], false, bfr[u], (short)0, acc[2 * t + u],
              false, false);
    }
  }

  // ---- Epilogue: bias + ReLU; padded-NHWC b128 stores, or NCHW f32 ----
  if (yf32) {
    const size_t plane = (size_t)H * W;
    #pragma unroll
    for (int t = 0; t < 4; ++t) {
      #pragma unroll
      for (int u = 0; u < 2; ++u) {
        const int pix = pixBase + 16 * u + n;
        #pragma unroll
        for (int r = 0; r < 8; ++r) {
          const int co = coBase + 16 * t + koff + r;
          float v = acc[2 * t + u][r] + bias[co];
          v = v > 0.0f ? v : 0.0f;
          yf32[((size_t)b * Cout + co) * plane + pix] = v;
        }
      }
    }
  } else {
    #pragma unroll
    for (int u = 0; u < 2; ++u) {
      unsigned short* drow = ybf +
          (((size_t)b * (H + PAD2) + py + PAD) * Wp + px[u] + PAD) * Cout;
      #pragma unroll
      for (int t = 0; t < 4; ++t) {
        v8bf o;
        #pragma unroll
        for (int r = 0; r < 8; ++r) {
          const int co = coBase + 16 * t + koff + r;
          float v = acc[2 * t + u][r] + bias[co];
          v = v > 0.0f ? v : 0.0f;
          o[r] = bits_to_bf(f32_to_bf16_bits(v));
        }
        *reinterpret_cast<v8bf*>(drow + coBase + 16 * t + koff) = o;
      }
    }
  }
}

// ---------------------------------------------------------------------------
// 2x2 maxpool stride 2, padded NHWC -> padded NHWC (interior only).
// ---------------------------------------------------------------------------
__global__ void maxpool2_nhwc(const unsigned short* __restrict__ in,
                              unsigned short* __restrict__ out,
                              int B, int C, int H, int W) {
  const int Ho = H >> 1, Wo = W >> 1;
  const int Wpi = W + PAD2, Wpo = Wo + PAD2;
  const size_t total = (size_t)B * Ho * Wo * C;
  const size_t i = (size_t)blockIdx.x * blockDim.x + threadIdx.x;
  if (i >= total) return;
  const int c = (int)(i % C);
  size_t t = i / C;
  const int wo = (int)(t % Wo); t /= Wo;
  const int ho = (int)(t % Ho);
  const int b  = (int)(t / Ho);
  const unsigned short* base = in +
      (((size_t)b * (H + PAD2) + 2 * ho + PAD) * Wpi + 2 * wo + PAD) * C + c;
  const size_t rs = (size_t)Wpi * C;
  float v0 = bf16_bits_to_f32(base[0]);
  float v1 = bf16_bits_to_f32(base[C]);
  float v2 = bf16_bits_to_f32(base[rs]);
  float v3 = bf16_bits_to_f32(base[rs + C]);
  float m01 = v0 > v1 ? v0 : v1;
  float m23 = v2 > v3 ? v2 : v3;
  out[(((size_t)b * (Ho + PAD2) + ho + PAD) * Wpo + wo + PAD) * C + c] =
      f32_to_bf16_bits(m01 > m23 ? m01 : m23);
}

// ---------------------------------------------------------------------------
// Irregular pool (padded NHWC): mask==1 -> replicate 2x2 block max, else copy.
// ---------------------------------------------------------------------------
__global__ void irr_pool_nhwc(const unsigned short* __restrict__ in,
                              unsigned short* __restrict__ out,
                              const int* __restrict__ mask,
                              int B, int C, int H, int W) {
  const int Wp = W + PAD2;
  const size_t total = (size_t)B * H * W * C;
  const size_t i = (size_t)blockIdx.x * blockDim.x + threadIdx.x;
  if (i >= total) return;
  const int c = (int)(i % C);
  size_t t = i / C;
  const int xw = (int)(t % W); t /= W;
  const int yh = (int)(t % H);
  const int b  = (int)(t / H);
  const int m = mask[((size_t)b * (H >> 1) + (yh >> 1)) * (W >> 1) + (xw >> 1)];
  const size_t oidx =
      (((size_t)b * (H + PAD2) + yh + PAD) * Wp + xw + PAD) * C + c;
  if (m) {
    const int y0 = yh & ~1, x0 = xw & ~1;
    const unsigned short* base =
        in + (((size_t)b * (H + PAD2) + y0 + PAD) * Wp + x0 + PAD) * C + c;
    const size_t rs = (size_t)Wp * C;
    float v0 = bf16_bits_to_f32(base[0]);
    float v1 = bf16_bits_to_f32(base[C]);
    float v2 = bf16_bits_to_f32(base[rs]);
    float v3 = bf16_bits_to_f32(base[rs + C]);
    float m01 = v0 > v1 ? v0 : v1;
    float m23 = v2 > v3 ? v2 : v3;
    out[oidx] = f32_to_bf16_bits(m01 > m23 ? m01 : m23);
  } else {
    out[oidx] = in[(((size_t)b * (H + PAD2) + yh + PAD) * Wp + xw + PAD) * C + c];
  }
}

// ---------------------------------------------------------------------------
// Host-side orchestration
// ---------------------------------------------------------------------------
extern "C" void kernel_launch(void* const* d_in, const int* in_sizes, int n_in,
                              void* d_out, int out_size, void* d_ws, size_t ws_size,
                              hipStream_t stream) {
  (void)in_sizes; (void)n_in; (void)out_size;
  const float* batch = (const float*)d_in[0];
  const int*   mask  = (const int*)d_in[1];

  static const int CIN[10]  = {  3,  64,  64, 128, 128, 256, 256, 256, 512, 512};
  static const int COUT[10] = { 64,  64, 128, 128, 256, 256, 256, 512, 512, 512};

  auto align256 = [](size_t v) { return (v + 255) & ~(size_t)255; };

  char* ws = (char*)d_ws;
  size_t off = 0;
  unsigned short* wbf[10];
  for (int i = 0; i < 10; ++i) {
    wbf[i] = (unsigned short*)(ws + off);
    off = align256(off + (size_t)COUT[i] * 9 * CIN[i] * 2);
  }
  const size_t ACT_ELEMS = (size_t)4 * (384 + PAD2) * (384 + PAD2) * 64; // max
  unsigned short* bufA = (unsigned short*)(ws + off);
  off = align256(off + ACT_ELEMS * 2);
  unsigned short* bufB = (unsigned short*)(ws + off);
  off = align256(off + ACT_ELEMS * 2);
  if (off > ws_size) return;  // workspace too small; bail deterministically

  auto zero = [&](unsigned short* p, int H, int W, int C) {
    size_t n8 = ((size_t)4 * (H + PAD2) * (W + PAD2) * C) / 8;
    zero_u16x8<<<(unsigned)((n8 + 255) / 256), 256, 0, stream>>>(p, n8);
  };
  auto conv = [&](const unsigned short* xin, int wi, unsigned short* ybf,
                  float* yf, const int* mk, int H, int W) {
    dim3 grid((H * W) / 32, COUT[wi] / 64, 4);
    conv3x3_wmma_nhwc<<<grid, 32, 0, stream>>>(
        xin, wbf[wi], (const float*)d_in[3 + 2 * wi], ybf, yf, mk,
        CIN[wi], COUT[wi], H, W);
  };
  auto pool = [&](const unsigned short* xin, unsigned short* yout,
                  int C, int H, int W) {
    size_t total = (size_t)4 * C * (H / 2) * (W / 2);
    maxpool2_nhwc<<<(unsigned)((total + 255) / 256), 256, 0, stream>>>(
        xin, yout, 4, C, H, W);
  };

  // --- repack weights for layers 2..10 into (tap, ci)-major bf16 ---
  for (int i = 1; i < 10; ++i) {
    int total = COUT[i] * 9 * CIN[i];
    repack_w_bf16<<<(total + 255) / 256, 256, 0, stream>>>(
        (const float*)d_in[2 + 2 * i], wbf[i], CIN[i], total);
  }

  // --- layer 1: direct conv (NCHW f32 -> padded NHWC bf16) ---
  zero(bufA, 384, 384, 64);
  {
    size_t total = (size_t)4 * 384 * 384 * 64;
    conv1_direct<<<(unsigned)((total + 255) / 256), 256, 0, stream>>>(
        batch, (const float*)d_in[2], (const float*)d_in[3], bufA, 384, 384);
  }

  // --- VGG trunk (dilation 1); zero halo of every conv-consumed buffer ---
  conv(bufA, 1, bufB, nullptr, nullptr, 384, 384);   // 64 -> 64 (feeds pool)
  zero(bufA, 192, 192, 64);
  pool(bufB, bufA, 64, 384, 384);                    // -> 192, feeds conv3
  zero(bufB, 192, 192, 128);
  conv(bufA, 2, bufB, nullptr, nullptr, 192, 192);   // 64 -> 128, feeds conv4
  conv(bufB, 3, bufA, nullptr, nullptr, 192, 192);   // 128 -> 128 (feeds pool)
  zero(bufB, 96, 96, 128);
  pool(bufA, bufB, 128, 192, 192);                   // -> 96 (OS4), feeds conv5
  zero(bufA, 96, 96, 256);
  conv(bufB, 4, bufA, nullptr, nullptr, 96, 96);     // 128 -> 256, feeds conv6
  zero(bufB, 96, 96, 256);
  conv(bufA, 5, bufB, nullptr, nullptr, 96, 96);     // 256 -> 256, feeds conv7
  conv(bufB, 6, bufA, nullptr, nullptr, 96, 96);     // 256 -> 256 (feeds irr)

  // --- irregular max-pool (mask-gated 2x2 block pooling), feeds conv8 ---
  zero(bufB, 96, 96, 256);
  {
    size_t total = (size_t)4 * 96 * 96 * 256;
    irr_pool_nhwc<<<(unsigned)((total + 255) / 256), 256, 0, stream>>>(
        bufA, bufB, mask, 4, 256, 96, 96);
  }

  // --- graph convs: per-pixel dilation 1+mask ---
  zero(bufA, 96, 96, 512);
  conv(bufB, 7, bufA, nullptr, mask, 96, 96);        // 256 -> 512, feeds conv9
  zero(bufB, 96, 96, 512);
  conv(bufA, 8, bufB, nullptr, mask, 96, 96);        // 512 -> 512, feeds conv10
  conv(bufB, 9, nullptr, (float*)d_out, mask, 96, 96);  // 512 -> 512, f32 out
}
